// FeatherNet_5446018531620
// MI455X (gfx1250) — matrix operand order repaired and checked by reading, not compiled, not run on hardware
//
#include <hip/hip_runtime.h>
#include <hip/hip_bf16.h>
#include <stddef.h>

// ---------------- problem constants ----------------
#define D_IN   1024
#define D_H    4096
#define D_OUT  1024
#define BATCH  4096
#define SZN    2898      // SIZE_N
#define SZM    725       // SIZE_M
#define KP     736       // SZM padded to multiple of 32 (23*32)
#define MPAD   2944      // SZN padded to multiple of 128 (23*128)
#define N_W1   (D_H * D_IN)
#define N_B1   (D_H)
#define N_W2   (D_OUT * D_H)
#define N_B2   (D_OUT)

typedef __bf16 bf16_t;
typedef __attribute__((ext_vector_type(8)))  __bf16 v8bf;
typedef __attribute__((ext_vector_type(16))) __bf16 v16bf;
typedef __attribute__((ext_vector_type(8)))  float  v8f;

// ---------------- WMMA fragment loads (wave32 layouts per ISA 7.12.2) ------
// A fragment: 16x32 bf16, M = lane&15; lanes 0-15 hold K {0..7,16..23},
// lanes 16-31 hold K {8..15,24..31}  -> two 16B loads at +0 and +16 (elem).
__device__ __forceinline__ v16bf load_a_frag(const bf16_t* A, int lda,
                                             int row_base, int k0, int lane) {
    const int r   = row_base + (lane & 15);
    const int off = (lane >> 4) << 3;              // 0 or 8
    const bf16_t* p = A + (size_t)r * lda + k0 + off;
    v8bf lo = *reinterpret_cast<const v8bf*>(p);
    v8bf hi = *reinterpret_cast<const v8bf*>(p + 16);
    return __builtin_shufflevector(lo, hi, 0,1,2,3,4,5,6,7,8,9,10,11,12,13,14,15);
}

// B fragment: 32x16 bf16, N = lane&15; lanes 0-15 hold K {0..7,16..23},
// lanes 16-31 hold K {8..15,24..31} -> two 16B loads at +0 and +16 (elem).
// B[k][n] = W[n][k] (W row-major [N,K]) so this is a contiguous row read.
__device__ __forceinline__ v16bf load_b_frag(const bf16_t* B, int ldb,
                                             int col_base, int k0, int lane) {
    const int r   = col_base + (lane & 15);
    const int off = (lane >> 4) << 3;              // 0 or 8
    const bf16_t* p = B + (size_t)r * ldb + k0 + off;
    v8bf lo = *reinterpret_cast<const v8bf*>(p);
    v8bf hi = *reinterpret_cast<const v8bf*>(p + 16);
    return __builtin_shufflevector(lo, hi, 0,1,2,3,4,5,6,7,8,9,10,11,12,13,14,15);
}

// D/C tile store: VGPR r -> M = (lane/16)*8 + r, N = lane&15.
template<bool RELU, bool OUTBF>
__device__ __forceinline__ void store_tile(const v8f& c, void* C, const float* bias,
                                           int M, int N, int ldc,
                                           int tm, int tn, int lane) {
    const int n  = tn + (lane & 15);
    const int mb = tm + ((lane >> 4) << 3);
    if (n >= N) return;
    const float bv = bias ? bias[n] : 0.0f;
#pragma unroll
    for (int r = 0; r < 8; ++r) {
        const int m = mb + r;
        if (m < M) {
            float v = c[r] + bv;
            if (RELU) v = v > 0.0f ? v : 0.0f;
            if (OUTBF) ((bf16_t*)C)[(size_t)m * ldc + n] = (bf16_t)v;
            else       ((float*)C)[(size_t)m * ldc + n]  = v;
        }
    }
}

// ---------------- generic GEMM: C[M,N] = A[M,K] * B[N,K]^T (+bias)(ReLU) ---
// block = (32,4): 4 waves arranged 2x2 -> 128x128 block tile.
// Each wave owns a 64x64 tile: 4x4 accumulators, 16 WMMA per K-step against
// 16 b128 loads (1.0 loads/WMMA) for maximal matrix-pipe utilization.
template<bool RELU, bool OUTBF>
__global__ __launch_bounds__(128)
void gemm_bt_wmma(const bf16_t* __restrict__ A, const bf16_t* __restrict__ B,
                  const float* __restrict__ bias, void* __restrict__ C,
                  int M, int N, int K, int ldc) {
    const int lane = threadIdx.x;                   // wave32 lane
    const int wave = threadIdx.y;                   // 0..3
    const int m0 = blockIdx.y * 128 + (wave >> 1) * 64;
    const int n0 = blockIdx.x * 128 + (wave & 1) * 64;

    v8f acc[4][4] = {};

    for (int k0 = 0; k0 < K; k0 += 32) {
        if (k0 + 32 < K) {   // warm next K-slab into cache (global_prefetch_b8)
            __builtin_prefetch(A + (size_t)(m0 + (lane & 15)) * K + k0 + 32, 0, 3);
            __builtin_prefetch(B + (size_t)(n0 + (lane & 15)) * K + k0 + 32, 0, 3);
        }
        v16bf af[4], bf[4];
#pragma unroll
        for (int i = 0; i < 4; ++i)
            af[i] = load_a_frag(A, K, m0 + i * 16, k0, lane);
#pragma unroll
        for (int j = 0; j < 4; ++j)
            bf[j] = load_b_frag(B, K, n0 + j * 16, k0, lane);
#pragma unroll
        for (int i = 0; i < 4; ++i)
#pragma unroll
            for (int j = 0; j < 4; ++j)
                acc[i][j] = __builtin_amdgcn_wmma_f32_16x16x32_bf16(
                    false, af[i], false, bf[j], (short)0, acc[i][j], false, false);
    }

#pragma unroll
    for (int i = 0; i < 4; ++i)
#pragma unroll
        for (int j = 0; j < 4; ++j)
            store_tile<RELU, OUTBF>(acc[i][j], C, bias, M, N, ldc,
                                    m0 + i * 16, n0 + j * 16, lane);
}

// ---------------- packing / conversion kernels -----------------------------
// V1 f32 [SZN,SZM] -> bf16 [MPAD,KP], zero padded.
__global__ void cvt_pad_v1(const float* __restrict__ V1, bf16_t* __restrict__ out) {
    int i = blockIdx.x * blockDim.x + threadIdx.x;
    if (i >= MPAD * KP) return;
    int r = i / KP, c = i - r * KP;
    out[i] = (r < SZN && c < SZM) ? (bf16_t)V1[r * SZM + c] : (bf16_t)0.0f;
}

// V2 f32 [SZM,SZN] -> transposed bf16 [MPAD,KP] (out[n,k] = V2[k,n]), padded.
__global__ void cvt_pad_v2t(const float* __restrict__ V2, bf16_t* __restrict__ out) {
    int i = blockIdx.x * blockDim.x + threadIdx.x;
    if (i >= MPAD * KP) return;
    int n = i / KP, k = i - n * KP;
    out[i] = (n < SZN && k < SZM) ? (bf16_t)V2[(size_t)k * SZN + n] : (bf16_t)0.0f;
}

// x f32 -> bf16
__global__ void cvt_x(const float* __restrict__ x, bf16_t* __restrict__ out, int n) {
    int i = blockIdx.x * blockDim.x + threadIdx.x;
    if (i < n) out[i] = (bf16_t)x[i];
}

// Slice vflat (row-major V, ld = SZN) into scaled bf16 weights + f32 biases.
__global__ void pack_params(const float* __restrict__ vflat,
                            const float* __restrict__ w1p, const float* __restrict__ b1p,
                            const float* __restrict__ w2p, const float* __restrict__ b2p,
                            bf16_t* __restrict__ W1, float* __restrict__ b1,
                            bf16_t* __restrict__ W2, float* __restrict__ b2) {
    int i = blockIdx.x * blockDim.x + threadIdx.x;
    const int total = N_W1 + N_B1 + N_W2 + N_B2;
    if (i >= total) return;
    float v = vflat[i];
    if (i < N_W1) {
        W1[i] = (bf16_t)(w1p[0] * v);
    } else if (i < N_W1 + N_B1) {
        b1[i - N_W1] = b1p[0] * v;
    } else if (i < N_W1 + N_B1 + N_W2) {
        W2[i - (N_W1 + N_B1)] = (bf16_t)(w2p[0] * v);
    } else {
        b2[i - (N_W1 + N_B1 + N_W2)] = b2p[0] * v;
    }
}

// ---------------- host launch ----------------------------------------------
static inline size_t align256(size_t x) { return (x + 255) & ~(size_t)255; }

extern "C" void kernel_launch(void* const* d_in, const int* in_sizes, int n_in,
                              void* d_out, int out_size, void* d_ws, size_t ws_size,
                              hipStream_t stream) {
    const float* x   = (const float*)d_in[0];
    const float* V1  = (const float*)d_in[1];
    const float* V2  = (const float*)d_in[2];
    const float* w1p = (const float*)d_in[3];
    const float* b1p = (const float*)d_in[4];
    const float* w2p = (const float*)d_in[5];
    const float* b2p = (const float*)d_in[6];
    float* y = (float*)d_out;

    // workspace carve-up
    char* base = (char*)d_ws;
    size_t off = 0;
    bf16_t* v1bf = (bf16_t*)(base + off); off = align256(off + (size_t)MPAD * KP * 2);
    bf16_t* v2t  = (bf16_t*)(base + off); off = align256(off + (size_t)MPAD * KP * 2);
    float*  vout = (float*) (base + off); off = align256(off + (size_t)SZN * SZN * 4);
    bf16_t* xbf  = (bf16_t*)(base + off); off = align256(off + (size_t)BATCH * D_IN * 2);
    bf16_t* W1   = (bf16_t*)(base + off); off = align256(off + (size_t)D_H * D_IN * 2);
    bf16_t* W2   = (bf16_t*)(base + off); off = align256(off + (size_t)D_OUT * D_H * 2);
    float*  b1   = (float*) (base + off); off = align256(off + (size_t)D_H * 4);
    float*  b2   = (float*) (base + off); off = align256(off + (size_t)D_OUT * 4);
    bf16_t* hbf  = (bf16_t*)(base + off); off = align256(off + (size_t)BATCH * D_H * 2);
    (void)ws_size;

    const dim3 cvtBlk(256);
    // stage inputs to bf16 (padded / transposed)
    cvt_pad_v1 <<<dim3((MPAD * KP + 255) / 256), cvtBlk, 0, stream>>>(V1, v1bf);
    cvt_pad_v2t<<<dim3((MPAD * KP + 255) / 256), cvtBlk, 0, stream>>>(V2, v2t);
    cvt_x      <<<dim3((BATCH * D_IN + 255) / 256), cvtBlk, 0, stream>>>(x, xbf, BATCH * D_IN);

    const dim3 gblk(32, 4);   // 4 wave32s, 128x128 block tile
    // V = V1 @ V2  (B operand is V2^T so both are row-major-in-K)
    gemm_bt_wmma<false, false><<<dim3(MPAD / 128, MPAD / 128), gblk, 0, stream>>>(
        v1bf, v2t, nullptr, vout, SZN, SZN, KP, SZN);

    // slice vflat -> scaled weights / biases
    const int ptot = N_W1 + N_B1 + N_W2 + N_B2;
    pack_params<<<dim3((ptot + 255) / 256), cvtBlk, 0, stream>>>(
        vout, w1p, b1p, w2p, b2p, W1, b1, W2, b2);

    // h = relu(x @ W1^T + b1), bf16 out
    gemm_bt_wmma<true, true><<<dim3(D_H / 128, BATCH / 128), gblk, 0, stream>>>(
        xbf, W1, b1, hbf, BATCH, D_H, D_IN, D_H);

    // y = h @ W2^T + b2, f32 out
    gemm_bt_wmma<false, false><<<dim3(D_OUT / 128, BATCH / 128), gblk, 0, stream>>>(
        hbf, W2, b2, y, BATCH, D_OUT, D_H, D_OUT);

    (void)in_sizes; (void)n_in; (void)out_size;
}
